// BoxTransformer_6399501271174
// MI455X (gfx1250) — matrix operand order repaired and checked by reading, not compiled, not run on hardware
//
#include <hip/hip_runtime.h>
#include <hip/hip_bf16.h>
#include <math.h>

// ---------------- problem constants ----------------
#define BB   16
#define PP   512
#define TOT  6400
#define DD   256
#define HH   8
#define ZZ   64
#define LL   4
#define HSZ  32
#define NN   513            // P + 1 (CLS)
#define BN   (BB*NN)        // 8208
#define D2   (2*DD)
#define ITILES ((NN+15)/16) // 33
#define NEGV (-1.0e30f)

typedef __attribute__((ext_vector_type(16))) _Float16 v16h;
typedef __attribute__((ext_vector_type(8)))  float    v8f;

union HFrag {
  v16h    v;
  _Float16 h[16];
  float4  f4[2];
};

__device__ __forceinline__ v8f wmma_f16(v16h a, v16h b, v8f c) {
  // D = A(16x32,f16) * B(32x16,f16) + C(16x16,f32)
  return __builtin_amdgcn_wmma_f32_16x16x32_f16(false, a, false, b, (short)0, c,
                                                false, false);
}

// ---------------- setup kernels ----------------

// x[b,0,:] = empty_embed; x[b,1+p,:] = embeds[page_ids[b,p]] (JAX -1 wraps)
__global__ __launch_bounds__(256) void gather_x_kernel(
    const float* __restrict__ embeds, const float* __restrict__ empty_e,
    const int* __restrict__ page_ids, float* __restrict__ x,
    _Float16* __restrict__ xh)
{
  int idx = blockIdx.x * 256 + threadIdx.x;
  if (idx >= BN * DD) return;
  int d   = idx & (DD - 1);
  int row = idx >> 8;           // DD == 256
  int n   = row % NN;
  int b   = row / NN;
  float v;
  if (n == 0) {
    v = empty_e[d];
  } else {
    int pid = page_ids[b * PP + n - 1];
    if (pid < 0) pid += TOT;    // JAX negative-index wrap
    v = embeds[(size_t)pid * DD + d];
  }
  x[idx]  = v;
  xh[idx] = (_Float16)v;
}

__global__ __launch_bounds__(256) void boxes_kernel(
    const float* __restrict__ xmin, const float* __restrict__ xmax,
    const float* __restrict__ ymin, const float* __restrict__ ymax,
    const float* __restrict__ width, const float* __restrict__ height,
    const int* __restrict__ page_ids,
    float* __restrict__ cxg, float* __restrict__ cyg,
    float* __restrict__ wg, float* __restrict__ hg)
{
  int idx = blockIdx.x * 256 + threadIdx.x;
  if (idx >= BB * PP) return;
  int pid = page_ids[idx];
  if (pid < 0) pid += TOT;
  cxg[idx] = 0.5f * (xmin[pid] + xmax[pid]);
  cyg[idx] = 0.5f * (ymin[pid] + ymax[pid]);
  wg[idx]  = width[pid];
  hg[idx]  = height[pid];
}

// rel[b,i,j] = (clip(round((cx_j-cx_i)/max(w_i,1e-3)),-32,31)+32)  | (...y...)<<8
__global__ __launch_bounds__(256) void rel_kernel(
    const float* __restrict__ cxg, const float* __restrict__ cyg,
    const float* __restrict__ wg, const float* __restrict__ hg,
    unsigned short* __restrict__ rel)
{
  size_t idx = (size_t)blockIdx.x * 256 + threadIdx.x;
  if (idx >= (size_t)BB * PP * PP) return;
  int j = (int)(idx % PP);
  int i = (int)((idx / PP) % PP);
  int b = (int)(idx / ((size_t)PP * PP));
  float dx = rintf((cxg[b*PP + j] - cxg[b*PP + i]) / fmaxf(wg[b*PP + i], 1e-3f));
  float dy = rintf((cyg[b*PP + j] - cyg[b*PP + i]) / fmaxf(hg[b*PP + i], 1e-3f));
  int zx = (int)fminf(fmaxf(dx, -32.f), 31.f) + 32;
  int zy = (int)fminf(fmaxf(dy, -32.f), 31.f) + 32;
  rel[idx] = (unsigned short)(zx | (zy << 8));
}

// W[K,N] fp32 -> WT[N,K] f16 (so WMMA B-fragments are contiguous 32B loads)
__global__ __launch_bounds__(256) void wtrans_kernel(
    const float* __restrict__ src, _Float16* __restrict__ dst, int K, int N)
{
  int idx = blockIdx.x * 256 + threadIdx.x;
  if (idx >= K * N) return;
  int k = idx / N, n = idx % N;
  dst[(size_t)n * K + k] = (_Float16)src[idx];
}

__global__ __launch_bounds__(256) void cvt_kernel(
    const float* __restrict__ src, _Float16* __restrict__ dst, int n)
{
  int idx = blockIdx.x * 256 + threadIdx.x;
  if (idx < n) dst[idx] = (_Float16)src[idx];
}

// ---------------- WMMA GEMM:  Y = act(X[M,K]f16 @ W[K,N] + bias) ----------------
// block = 128 threads = 4 waves; each wave computes 16(M)x64(N) with FOUR
// independent accumulators (A-frag reused 4x, no WMMA->WMMA RAW chain);
// block tile = 16 x 256.
__global__ __launch_bounds__(128) void gemm_kernel(
    const _Float16* __restrict__ X, const _Float16* __restrict__ WT,
    const float* __restrict__ bias, float* __restrict__ Yf32,
    _Float16* __restrict__ Yf16, int M, int K, int Nout, int act)
{
  int wave = threadIdx.x >> 5;
  int lane = threadIdx.x & 31;
  int lhi  = lane >> 4, llo = lane & 15;
  int n0   = blockIdx.x * 256 + wave * 64;
  int m0   = blockIdx.y * 16;
  if (n0 >= Nout) return;                       // uniform per wave
  int row  = m0 + llo; if (row >= M) row = M - 1;
  const _Float16* xrow = X  + (size_t)row * K + lhi * 16;         // A frag base
  const _Float16* wrow = WT + (size_t)(n0 + llo) * K + lhi * 16;  // B frag base
  const size_t wstep = (size_t)16 * K;          // next 16-column group of WT
  v8f acc0 = {0.f,0.f,0.f,0.f,0.f,0.f,0.f,0.f};
  v8f acc1 = acc0, acc2 = acc0, acc3 = acc0;
  for (int kk = 0; kk < K; kk += 32) {
    HFrag a, b0, b1, b2, b3;
    // A 16x32 f16: lanes 0-15 rows, K 0..15; lanes 16-31 K 16..31
    a.f4[0] = *(const float4*)(xrow + kk);
    a.f4[1] = *(const float4*)(xrow + kk + 8);
    // B 32x16 f16: lane column n=llo, 16 contiguous K values per half-wave
    b0.f4[0] = *(const float4*)(wrow + kk);
    b0.f4[1] = *(const float4*)(wrow + kk + 8);
    b1.f4[0] = *(const float4*)(wrow + wstep + kk);
    b1.f4[1] = *(const float4*)(wrow + wstep + kk + 8);
    b2.f4[0] = *(const float4*)(wrow + 2 * wstep + kk);
    b2.f4[1] = *(const float4*)(wrow + 2 * wstep + kk + 8);
    b3.f4[0] = *(const float4*)(wrow + 3 * wstep + kk);
    b3.f4[1] = *(const float4*)(wrow + 3 * wstep + kk + 8);
    if (kk + 32 < K) __builtin_prefetch(xrow + kk + 64, 0, 3);  // WGP-scope
    acc0 = wmma_f16(a.v, b0.v, acc0);   // 4 independent XDL ops per k-step
    acc1 = wmma_f16(a.v, b1.v, acc1);
    acc2 = wmma_f16(a.v, b2.v, acc2);
    acc3 = wmma_f16(a.v, b3.v, acc3);
  }
  v8f accs[4] = {acc0, acc1, acc2, acc3};
  #pragma unroll
  for (int t = 0; t < 4; t++) {
    int nc = n0 + t * 16 + llo;
    float bn = bias[nc];
    #pragma unroll
    for (int v = 0; v < 8; v++) {
      int m = m0 + v + lhi * 8;         // C layout: row = v + 8*(lane>>4)
      if (m >= M) continue;
      float val = accs[t][v] + bn;
      if (act == 1) val = 0.5f * val * (1.0f + erff(val * 0.70710678118f)); // exact GELU
      size_t o = (size_t)m * Nout + nc;
      if (Yf32) Yf32[o] = val;
      if (Yf16) Yf16[o] = (_Float16)val;
    }
  }
}

// ---------------- positional projection tables ----------------
// qpk[r,h,z,c] = sum_d q[r, h*32+c*16+d] * pk[z, c*128+h*16+d]
__global__ __launch_bounds__(256) void posproj_kernel(
    const _Float16* __restrict__ xp,   // q or k, [BN, DD] f16
    const _Float16* __restrict__ pe,   // pk or pq, [ZZ, DD] f16
    float* __restrict__ outp)          // [BN, HH, ZZ, 2] f32
{
  int idx = blockIdx.x * 256 + threadIdx.x;
  if (idx >= BN * HH * ZZ) return;
  int z = idx % ZZ;
  int h = (idx / ZZ) % HH;
  int r = idx / (ZZ * HH);
  #pragma unroll
  for (int c = 0; c < 2; c++) {
    const _Float16* qv = xp + (size_t)r * DD + h * HSZ + c * 16;
    const _Float16* pv = pe + (size_t)z * DD + c * 128 + h * 16;
    float s = 0.f;
    #pragma unroll
    for (int d = 0; d < 16; d++) s += (float)qv[d] * (float)pv[d];
    outp[(((size_t)r * HH + h) * ZZ + z) * 2 + c] = s;
  }
}

// ---------------- flash attention with disentangled positional bias ----------------
// 1 wave per (b, h, 16-row i-tile). WMMA for QK^T and P*V, online softmax.
__global__ __launch_bounds__(32) void attn_kernel(
    const _Float16* __restrict__ qh, const _Float16* __restrict__ kh,
    const _Float16* __restrict__ vh, const float* __restrict__ qpk,
    const float* __restrict__ kpq, const unsigned short* __restrict__ rel,
    const int* __restrict__ page_ids, _Float16* __restrict__ ctx)
{
  int bid = blockIdx.x;
  int it  = bid % ITILES;
  int tmp = bid / ITILES;
  int h   = tmp % HH;
  int b   = tmp / HH;
  int i0  = it * 16;
  int lane = threadIdx.x;
  int lhi  = lane >> 4, llo = lane & 15;
  const float scale = 0.17677669529663687f; // 1/sqrt(32)

  __shared__ __align__(16) float    qpk_s[16][ZZ * 2]; // 8 KB: qpk slice for head h
  __shared__ __align__(16) _Float16 p_s[16][32];       // probs staging (transpose)
  __shared__ __align__(16) _Float16 v_s[32][32];       // V tile staging

  // stage qpk for rows i0..i0+15, head h
  for (int t = lane; t < 16 * ZZ * 2; t += 32) {
    int r  = t >> 7;          // ZZ*2 == 128
    int zc = t & 127;
    int gi = i0 + r; if (gi >= NN) gi = NN - 1;
    qpk_s[r][zc] = qpk[(((size_t)(b * NN + gi) * HH + h) * ZZ) * 2 + zc];
  }
  // load Q A-fragment (16 rows x 32 HS)
  HFrag qa;
  {
    int gi = i0 + llo; if (gi >= NN) gi = NN - 1;
    const _Float16* qrow = qh + (size_t)(b * NN + gi) * DD + h * HSZ + lhi * 16;
    qa.f4[0] = *(const float4*)(qrow);
    qa.f4[1] = *(const float4*)(qrow + 8);
  }
  __syncthreads();

  float mrow[8], srow[8];
  #pragma unroll
  for (int v = 0; v < 8; v++) { mrow[v] = NEGV; srow[v] = 0.f; }
  v8f o0 = {0.f,0.f,0.f,0.f,0.f,0.f,0.f,0.f};
  v8f o1 = {0.f,0.f,0.f,0.f,0.f,0.f,0.f,0.f};

  for (int j0 = 0; j0 < NN; j0 += 32) {
    // ---- QK^T: two 16x16 score tiles via WMMA ----
    HFrag kb0, kb1;
    {
      int ja = j0 + llo;      if (ja >= NN) ja = NN - 1;
      int jb = j0 + 16 + llo; if (jb >= NN) jb = NN - 1;
      const _Float16* ka = kh + (size_t)(b * NN + ja) * DD + h * HSZ + lhi * 16;
      const _Float16* kb = kh + (size_t)(b * NN + jb) * DD + h * HSZ + lhi * 16;
      kb0.f4[0] = *(const float4*)(ka);
      kb0.f4[1] = *(const float4*)(ka + 8);
      kb1.f4[0] = *(const float4*)(kb);
      kb1.f4[1] = *(const float4*)(kb + 8);
    }
    v8f z8 = {0.f,0.f,0.f,0.f,0.f,0.f,0.f,0.f};
    v8f s0 = wmma_f16(qa.v, kb0.v, z8);
    v8f s1 = wmma_f16(qa.v, kb1.v, z8);

    // ---- positional bias + mask + online softmax ----
    int jA = j0 + llo, jB = j0 + 16 + llo;
    bool validA = (jA < NN) && (jA == 0 || page_ids[b * PP + jA - 1] != -1);
    bool validB = (jB < NN) && (jB == 0 || page_ids[b * PP + jB - 1] != -1);
    float l0[8], l1[8];
    #pragma unroll
    for (int v = 0; v < 8; v++) {
      int m  = v + lhi * 8;
      int gi = i0 + m;
      float pos0 = 0.f, pos1 = 0.f;
      if (gi >= 1 && gi < NN) {
        if (jA >= 1 && jA < NN) {
          unsigned r16 = rel[((size_t)b * PP + (gi - 1)) * PP + (jA - 1)];
          int zx = r16 & 63, zy = (r16 >> 8) & 63;
          size_t kb_ = ((size_t)(b * NN + jA) * HH + h) * ZZ;
          pos0 = qpk_s[m][zx * 2] + qpk_s[m][zy * 2 + 1]
               + kpq[(kb_ + zx) * 2] + kpq[(kb_ + zy) * 2 + 1];
        }
        if (jB >= 1 && jB < NN) {
          unsigned r16 = rel[((size_t)b * PP + (gi - 1)) * PP + (jB - 1)];
          int zx = r16 & 63, zy = (r16 >> 8) & 63;
          size_t kb_ = ((size_t)(b * NN + jB) * HH + h) * ZZ;
          pos1 = qpk_s[m][zx * 2] + qpk_s[m][zy * 2 + 1]
               + kpq[(kb_ + zx) * 2] + kpq[(kb_ + zy) * 2 + 1];
        }
      }
      l0[v] = validA ? (s0[v] + pos0) * scale : NEGV;
      l1[v] = validB ? (s1[v] + pos1) * scale : NEGV;
    }
    #pragma unroll
    for (int v = 0; v < 8; v++) {
      int m = v + lhi * 8;
      float mx = fmaxf(l0[v], l1[v]);
      #pragma unroll
      for (int off = 1; off < 16; off <<= 1) mx = fmaxf(mx, __shfl_xor(mx, off, 32));
      float nm   = fmaxf(mrow[v], mx);
      float corr = __expf(mrow[v] - nm);
      float p0 = __expf(l0[v] - nm);
      float p1 = __expf(l1[v] - nm);
      float rs = p0 + p1;
      #pragma unroll
      for (int off = 1; off < 16; off <<= 1) rs += __shfl_xor(rs, off, 32);
      srow[v] = srow[v] * corr + rs;
      mrow[v] = nm;
      o0[v] *= corr;
      o1[v] *= corr;
      p_s[m][llo]      = (_Float16)p0;
      p_s[m][16 + llo] = (_Float16)p1;
    }
    // ---- stage V tile (32 keys x 16 dims) ----
    {
      int j = j0 + lane; if (j >= NN) j = NN - 1;
      const float4* vr = (const float4*)(vh + (size_t)(b * NN + j) * DD + h * HSZ);
      float4* dst = (float4*)(&v_s[lane][0]);
      dst[0] = vr[0];
      dst[1] = vr[1];
    }
    __syncthreads();
    // ---- P * V via WMMA (K = 32 keys) ----
    HFrag pa, vb0, vb1;
    {
      const _Float16* prow = &p_s[llo][0] + lhi * 16;
      pa.f4[0] = *(const float4*)(prow);
      pa.f4[1] = *(const float4*)(prow + 8);
    }
    #pragma unroll
    for (int e = 0; e < 16; e++) {
      int jl = lhi * 16 + e;
      vb0.h[e] = v_s[jl][llo];
      vb1.h[e] = v_s[jl][16 + llo];
    }
    o0 = wmma_f16(pa.v, vb0.v, o0);
    o1 = wmma_f16(pa.v, vb1.v, o1);
    __syncthreads();
  }

  // finalize: divide by softmax sum, store f16 context
  #pragma unroll
  for (int v = 0; v < 8; v++) {
    int m  = v + lhi * 8;
    int gi = i0 + m;
    if (gi >= NN) continue;
    float inv = 1.0f / srow[v];
    size_t off = (size_t)(b * NN + gi) * DD + h * HSZ;
    ctx[off + llo]      = (_Float16)(o0[v] * inv);
    ctx[off + 16 + llo] = (_Float16)(o1[v] * inv);
  }
}

// ---------------- residual + layernorm (one row per block) ----------------
__global__ __launch_bounds__(256) void resln_kernel(
    float* __restrict__ x, _Float16* __restrict__ xh,
    const float* __restrict__ upd, const float* __restrict__ res,
    const float* __restrict__ gamma, const float* __restrict__ beta)
{
  __shared__ float red[256];
  int r = blockIdx.x, t = threadIdx.x;
  size_t o = (size_t)r * DD + t;
  float y = x[o] + upd[o] * res[0];
  red[t] = y; __syncthreads();
  for (int s = 128; s > 0; s >>= 1) { if (t < s) red[t] += red[t + s]; __syncthreads(); }
  float mean = red[0] * (1.0f / DD);
  __syncthreads();
  float d = y - mean;
  red[t] = d * d; __syncthreads();
  for (int s = 128; s > 0; s >>= 1) { if (t < s) red[t] += red[t + s]; __syncthreads(); }
  float var = red[0] * (1.0f / DD);
  float ov = d * rsqrtf(var + 1e-5f) * gamma[t] + beta[t];
  x[o]  = ov;
  xh[o] = (_Float16)ov;
}

// ---------------- final: drop CLS, zero padded boxes ----------------
__global__ __launch_bounds__(256) void final_kernel(
    const float* __restrict__ x, const int* __restrict__ page_ids,
    float* __restrict__ out)
{
  int idx = blockIdx.x * 256 + threadIdx.x;
  if (idx >= BB * PP * DD) return;
  int d = idx & (DD - 1);
  int p = (idx >> 8) % PP;
  int b = idx / (PP * DD);
  bool valid = page_ids[b * PP + p] != -1;
  out[idx] = valid ? x[(size_t)(b * NN + p + 1) * DD + d] : 0.f;
}

// ---------------- host orchestration ----------------
extern "C" void kernel_launch(void* const* d_in, const int* in_sizes, int n_in,
                              void* d_out, int out_size, void* d_ws, size_t ws_size,
                              hipStream_t stream)
{
  (void)in_sizes; (void)n_in; (void)out_size; (void)ws_size;
  const float* embeds  = (const float*)d_in[0];
  const float* xmin    = (const float*)d_in[1];
  const float* xmax    = (const float*)d_in[2];
  const float* ymin    = (const float*)d_in[3];
  const float* ymax    = (const float*)d_in[4];
  const float* width   = (const float*)d_in[5];
  const float* height  = (const float*)d_in[6];
  const float* empty_e = (const float*)d_in[7];
  const float* posemb  = (const float*)d_in[8];
  const float* cq_w = (const float*)d_in[9];  const float* cq_b = (const float*)d_in[10];
  const float* ck_w = (const float*)d_in[11]; const float* ck_b = (const float*)d_in[12];
  const float* pk_w = (const float*)d_in[13]; const float* pk_b = (const float*)d_in[14];
  const float* pq_w = (const float*)d_in[15]; const float* pq_b = (const float*)d_in[16];
  const float* cv_w = (const float*)d_in[17]; const float* cv_b = (const float*)d_in[18];
  const float* out_w = (const float*)d_in[19]; const float* out_b = (const float*)d_in[20];
  const float* res1 = (const float*)d_in[21]; const float* res2 = (const float*)d_in[22];
  const float* ln1_g = (const float*)d_in[23]; const float* ln1_b = (const float*)d_in[24];
  const float* ln2_g = (const float*)d_in[25]; const float* ln2_b = (const float*)d_in[26];
  const float* ff1_w = (const float*)d_in[27]; const float* ff1_b = (const float*)d_in[28];
  const float* ff2_w = (const float*)d_in[29]; const float* ff2_b = (const float*)d_in[30];
  const int* page_ids = (const int*)d_in[31];

  char* ws = (char*)d_ws;
  size_t off = 0;
  auto alloc = [&](size_t bytes) -> void* {
    void* p = ws + off;
    off = (off + bytes + 255) & ~(size_t)255;
    return p;
  };

  float*    x    = (float*)   alloc((size_t)BN * DD * 4);
  _Float16* xh   = (_Float16*)alloc((size_t)BN * DD * 2);
  _Float16* qh   = (_Float16*)alloc((size_t)BN * DD * 2);
  _Float16* khb  = (_Float16*)alloc((size_t)BN * DD * 2);
  _Float16* vhb  = (_Float16*)alloc((size_t)BN * DD * 2);
  _Float16* ctx  = (_Float16*)alloc((size_t)BN * DD * 2);
  _Float16* h1   = (_Float16*)alloc((size_t)BN * D2 * 2);
  float*    gout = (float*)   alloc((size_t)BN * DD * 4);
  float*    qpkb = (float*)   alloc((size_t)BN * HH * ZZ * 2 * 4);
  float*    kpqb = (float*)   alloc((size_t)BN * HH * ZZ * 2 * 4);
  unsigned short* relb = (unsigned short*)alloc((size_t)BB * PP * PP * 2);
  float* cxg = (float*)alloc((size_t)BB * PP * 4);
  float* cyg = (float*)alloc((size_t)BB * PP * 4);
  float* wgb = (float*)alloc((size_t)BB * PP * 4);
  float* hgb = (float*)alloc((size_t)BB * PP * 4);
  _Float16* pkh  = (_Float16*)alloc((size_t)ZZ * DD * 2);
  _Float16* pqh  = (_Float16*)alloc((size_t)ZZ * DD * 2);
  _Float16* posh = (_Float16*)alloc((size_t)ZZ * DD * 2);
  // transposed f16 weights: per layer cq,ck,cv,out,pkw,pqw (DD*DD) + ff1,ff2 (DD*D2)
  const size_t WSQ = (size_t)DD * DD;       // 65536
  const size_t WFF = (size_t)DD * D2;       // 131072
  const size_t WPL = 6 * WSQ + 2 * WFF;     // per-layer halves
  _Float16* wt = (_Float16*)alloc(LL * WPL * 2);

  auto tr = [&](const float* src, _Float16* dst, int K, int N) {
    wtrans_kernel<<<(K * N + 255) / 256, 256, 0, stream>>>(src, dst, K, N);
  };

  // one-time prep
  for (int l = 0; l < LL; l++) {
    _Float16* w = wt + (size_t)l * WPL;
    tr(cq_w  + (size_t)l * WSQ, w,                         DD, DD);
    tr(ck_w  + (size_t)l * WSQ, w + WSQ,                   DD, DD);
    tr(cv_w  + (size_t)l * WSQ, w + 2 * WSQ,               DD, DD);
    tr(out_w + (size_t)l * WSQ, w + 3 * WSQ,               DD, DD);
    tr(pk_w  + (size_t)l * WSQ, w + 4 * WSQ,               DD, DD);
    tr(pq_w  + (size_t)l * WSQ, w + 5 * WSQ,               DD, DD);
    tr(ff1_w + (size_t)l * WFF, w + 6 * WSQ,               DD, D2);
    tr(ff2_w + (size_t)l * WFF, w + 6 * WSQ + WFF,         D2, DD);
  }
  cvt_kernel<<<(ZZ * DD + 255) / 256, 256, 0, stream>>>(posemb, posh, ZZ * DD);
  gather_x_kernel<<<(BN * DD + 255) / 256, 256, 0, stream>>>(
      embeds, empty_e, page_ids, x, xh);
  boxes_kernel<<<(BB * PP + 255) / 256, 256, 0, stream>>>(
      xmin, xmax, ymin, ymax, width, height, page_ids, cxg, cyg, wgb, hgb);
  {
    size_t tot = (size_t)BB * PP * PP;
    rel_kernel<<<(unsigned)((tot + 255) / 256), 256, 0, stream>>>(cxg, cyg, wgb, hgb, relb);
  }

  const dim3 gBlk(128);
  const dim3 gBig(1, BN / 16);           // Nout=256: one block covers all N
  const dim3 gFF1(2, BN / 16);           // Nout=512
  const dim3 gPos(1, ZZ / 16);           // [64,256]
  const int  posGrid = (BN * HH * ZZ + 255) / 256;

  for (int l = 0; l < LL; l++) {
    _Float16* w = wt + (size_t)l * WPL;
    // q, k, v projections (f16 out)
    gemm_kernel<<<gBig, gBlk, 0, stream>>>(xh, w,           cq_b + (size_t)l * DD,
                                           nullptr, qh,  BN, DD, DD, 0);
    gemm_kernel<<<gBig, gBlk, 0, stream>>>(xh, w + WSQ,     ck_b + (size_t)l * DD,
                                           nullptr, khb, BN, DD, DD, 0);
    gemm_kernel<<<gBig, gBlk, 0, stream>>>(xh, w + 2 * WSQ, cv_b + (size_t)l * DD,
                                           nullptr, vhb, BN, DD, DD, 0);
    // positional key/query projections (f16 out)
    gemm_kernel<<<gPos, gBlk, 0, stream>>>(posh, w + 4 * WSQ, pk_b + (size_t)l * DD,
                                           nullptr, pkh, ZZ, DD, DD, 0);
    gemm_kernel<<<gPos, gBlk, 0, stream>>>(posh, w + 5 * WSQ, pq_b + (size_t)l * DD,
                                           nullptr, pqh, ZZ, DD, DD, 0);
    // qpk / kpq lookup tables
    posproj_kernel<<<posGrid, 256, 0, stream>>>(qh,  pkh, qpkb);
    posproj_kernel<<<posGrid, 256, 0, stream>>>(khb, pqh, kpqb);
    // flash attention with disentangled bias
    attn_kernel<<<BB * HH * ITILES, 32, 0, stream>>>(
        qh, khb, vhb, qpkb, kpqb, relb, page_ids, ctx);
    // output projection (f32 out) + residual + LN1
    gemm_kernel<<<gBig, gBlk, 0, stream>>>(ctx, w + 3 * WSQ, out_b + (size_t)l * DD,
                                           gout, nullptr, BN, DD, DD, 0);
    resln_kernel<<<BN, 256, 0, stream>>>(x, xh, gout, res1 + l,
                                         ln1_g + (size_t)l * DD, ln1_b + (size_t)l * DD);
    // FFN: gelu(x@ff1+b) @ ff2 + b, residual + LN2
    gemm_kernel<<<gFF1, gBlk, 0, stream>>>(xh, w + 6 * WSQ, ff1_b + (size_t)l * D2,
                                           nullptr, h1, BN, DD, D2, 1);
    gemm_kernel<<<gBig, gBlk, 0, stream>>>(h1, w + 6 * WSQ + WFF, ff2_b + (size_t)l * DD,
                                           gout, nullptr, BN, D2, DD, 0);
    resln_kernel<<<BN, 256, 0, stream>>>(x, xh, gout, res2 + l,
                                         ln2_g + (size_t)l * DD, ln2_b + (size_t)l * DD);
  }

  final_kernel<<<(BB * PP * DD + 255) / 256, 256, 0, stream>>>(
      x, page_ids, (float*)d_out);
}
